// GINE_11218454577897
// MI455X (gfx1250) — compile-verified, matrix operand order checked by reading, not run on hardware
//
#include <hip/hip_runtime.h>
#include <hip/hip_bf16.h>

// ---------------------------------------------------------------------------
// GINE (2x GINEConv + MLP + mean-pool + linear head) for MI455X / gfx1250.
// - All dense GEMMs: v_wmma_f32_16x16x32_bf16 (wave32, 16x16 tiles).
// - Weights staged once per 256-thread block into LDS (amortized over 8 waves'
//   tiles); ONE tile per wave -> no loop for LICM to hoist B-fragments out of
//   (round-2 persistent loop caused scratch spills of hoisted fragments).
// - Edge GEMMs fused with gather(+x[src]) / relu / scatter-atomicAdd.
// - Node MLP fuses residual + GEMM1 + relu + LDS restage + GEMM2.
// ---------------------------------------------------------------------------

typedef __attribute__((ext_vector_type(16))) __bf16 v16bf;
typedef __attribute__((ext_vector_type(8)))  float  v8f;

#define N_NODES 50000
#define N_EDGES 800000
#define C_IN    64
#define C_HID   128
#define C_OUT   16
#define N_GRAPH 512

// native bf16 convert (lowers to v_cvt_pk_bf16_f32 on gfx1250)
__device__ __forceinline__ __bf16 f2bf(float f) { return (__bf16)f; }

// K index of element j in a 16-elem bf16 A/B fragment (ISA 7.12.2):
// lanes 0-15 hold K {0..7,16..23}, lanes 16-31 hold K {8..15,24..31}
__device__ __forceinline__ int kmap(int j, int half) {
  return (j < 8) ? (j + 8 * half) : (j + 8 + 8 * half);
}

__device__ __forceinline__ v8f wmma_bf16(v16bf a, v16bf b, v8f c) {
  return __builtin_amdgcn_wmma_f32_16x16x32_bf16(false, a, false, b,
                                                 (short)0, c, false, false);
}

// block-cooperative 16B-vector copy global(bf16) -> LDS(bf16); n multiple of 8
__device__ __forceinline__ void stage_bf16(const __bf16* __restrict__ g,
                                           __bf16* l, int n) {
  const uint4* gs = (const uint4*)g;
  uint4* ls = (uint4*)l;
  for (int i = threadIdx.x; i < (n >> 3); i += blockDim.x) ls[i] = gs[i];
}

// ---------------------------------------------------------------------------
__global__ void k_cvt_bf16(const float* __restrict__ s, __bf16* __restrict__ d, int n) {
  int i = blockIdx.x * blockDim.x + threadIdx.x;
  if (i < n) d[i] = f2bf(s[i]);
}

__global__ void k_zero(float* __restrict__ p, int n) {
  int i = blockIdx.x * blockDim.x + threadIdx.x;
  int stride = gridDim.x * blockDim.x;
  for (; i < n; i += stride) p[i] = 0.0f;
}

// ---------------------------------------------------------------------------
// e_proj[E,128] = bf16( edge_attr[E,64] @ We^T + be ).  One wave per 16 edges.
__global__ void k_edge_proj(const float* __restrict__ ea,    // [E][64]
                            const __bf16* __restrict__ Weg,  // [128][64] row-major
                            const float* __restrict__ be,    // [128]
                            __bf16* __restrict__ eproj,      // [E][128]
                            int tiles) {
  __shared__ __align__(16) __bf16 sW[C_HID * C_IN];          // 16 KB
  stage_bf16(Weg, sW, C_HID * C_IN);
  __syncthreads();

  int wave = threadIdx.x >> 5, lane = threadIdx.x & 31;
  int half = lane >> 4, l16 = lane & 15;
  int tile = blockIdx.x * (blockDim.x >> 5) + wave;
  if (tile >= tiles) return;
  int e0 = tile * 16;

  v8f acc[8] = {};
  const float* arow = ea + (size_t)(e0 + l16) * C_IN;
#pragma unroll
  for (int kk = 0; kk < 2; ++kk) {                 // K = 64 -> 2 steps of 32
    v16bf a;
#pragma unroll
    for (int j = 0; j < 16; ++j) a[j] = f2bf(arow[kk * 32 + kmap(j, half)]);
#pragma unroll
    for (int nt = 0; nt < 8; ++nt) {
      const __bf16* brow = sW + (nt * 16 + l16) * C_IN;
      v16bf b;
#pragma unroll
      for (int j = 0; j < 16; ++j) b[j] = brow[kk * 32 + kmap(j, half)];
      acc[nt] = wmma_bf16(a, b, acc[nt]);
    }
  }
#pragma unroll
  for (int nt = 0; nt < 8; ++nt) {
    int n = nt * 16 + l16;
    float bias = be[n];
#pragma unroll
    for (int r = 0; r < 8; ++r) {
      int m = r + 8 * half;
      eproj[(size_t)(e0 + m) * C_HID + n] = f2bf(acc[nt][r] + bias);
    }
  }
}

// ---------------------------------------------------------------------------
// Fused conv edge phase: e = e_proj @ Wc^T + bc; m = relu(x[src] + e);
// atomicAdd(aggr[dst], m).   DN = node feature dim (64 for conv1, 128 conv2).
template <int DN>
__global__ void k_conv_edge(const __bf16* __restrict__ eproj, // [E][128]
                            const __bf16* __restrict__ Wcg,   // [DN][128]
                            const float* __restrict__ bc,     // [DN]
                            const float* __restrict__ xin,    // [N][DN]
                            const int* __restrict__ src,
                            const int* __restrict__ dst,
                            float* __restrict__ aggr,         // [N][DN]
                            int tiles) {
  constexpr int NT = DN / 16;
  __shared__ __align__(16) __bf16 sW[DN * C_HID];             // 16 or 32 KB
  stage_bf16(Wcg, sW, DN * C_HID);
  __syncthreads();

  int wave = threadIdx.x >> 5, lane = threadIdx.x & 31;
  int half = lane >> 4, l16 = lane & 15;
  int tile = blockIdx.x * (blockDim.x >> 5) + wave;
  if (tile >= tiles) return;
  int e0 = tile * 16;

  v8f acc[NT] = {};
  const __bf16* arow = eproj + (size_t)(e0 + l16) * C_HID;
#pragma unroll
  for (int kk = 0; kk < 4; ++kk) {                 // K = 128 -> 4 steps
    v16bf a;
#pragma unroll
    for (int j = 0; j < 16; ++j) a[j] = arow[kk * 32 + kmap(j, half)];
#pragma unroll
    for (int nt = 0; nt < NT; ++nt) {
      const __bf16* brow = sW + (nt * 16 + l16) * C_HID;
      v16bf b;
#pragma unroll
      for (int j = 0; j < 16; ++j) b[j] = brow[kk * 32 + kmap(j, half)];
      acc[nt] = wmma_bf16(a, b, acc[nt]);
    }
  }

  int sE[8], dE[8];
#pragma unroll
  for (int r = 0; r < 8; ++r) {
    int e = e0 + r + 8 * half;
    sE[r] = src[e];
    dE[r] = dst[e];
  }
#pragma unroll
  for (int nt = 0; nt < NT; ++nt) {
    int n = nt * 16 + l16;
    float bias = bc[n];
#pragma unroll
    for (int r = 0; r < 8; ++r) {
      float v = acc[nt][r] + bias + xin[(size_t)sE[r] * DN + n];
      v = v > 0.0f ? v : 0.0f;
      atomicAdd(&aggr[(size_t)dE[r] * DN + n], v);
    }
  }
}

// ---------------------------------------------------------------------------
// Fused node phase: h = xin + aggr; t = relu(h @ W1^T + b1) (via LDS bf16);
// out = t @ W2^T + b2; optional final relu.  One wave per 16 nodes.
template <int DIN, bool RELU>
__global__ void k_mlp(const float* __restrict__ xin,   // [N][DIN]
                      const float* __restrict__ aggr,  // [N][DIN]
                      const __bf16* __restrict__ W1g,  // [128][DIN]
                      const float* __restrict__ b1,    // [128]
                      const __bf16* __restrict__ W2g,  // [128][128]
                      const float* __restrict__ b2,    // [128]
                      float* __restrict__ out,         // [N][128]
                      int tiles) {
  __shared__ __align__(16) __bf16 sW1[C_HID * DIN];           // 16 or 32 KB
  __shared__ __align__(16) __bf16 sW2[C_HID * C_HID];         // 32 KB
  __shared__ __align__(16) __bf16 sT[8 * 16 * C_HID];         // 8 waves x 4 KB
  stage_bf16(W1g, sW1, C_HID * DIN);
  stage_bf16(W2g, sW2, C_HID * C_HID);
  __syncthreads();

  int wave = threadIdx.x >> 5, lane = threadIdx.x & 31;
  int half = lane >> 4, l16 = lane & 15;
  int tile = blockIdx.x * (blockDim.x >> 5) + wave;
  if (tile >= tiles) return;
  int r0 = tile * 16;
  __bf16* T = sT + wave * (16 * C_HID);

  // ---- GEMM1: (x + aggr) @ W1^T ----
  v8f acc[8] = {};
  const float* xr = xin + (size_t)(r0 + l16) * DIN;
  const float* gr = aggr + (size_t)(r0 + l16) * DIN;
#pragma unroll
  for (int kk = 0; kk < DIN / 32; ++kk) {
    v16bf a;
#pragma unroll
    for (int j = 0; j < 16; ++j) {
      int k = kk * 32 + kmap(j, half);
      a[j] = f2bf(xr[k] + gr[k]);
    }
#pragma unroll
    for (int nt = 0; nt < 8; ++nt) {
      const __bf16* brow = sW1 + (nt * 16 + l16) * DIN;
      v16bf b;
#pragma unroll
      for (int j = 0; j < 16; ++j) b[j] = brow[kk * 32 + kmap(j, half)];
      acc[nt] = wmma_bf16(a, b, acc[nt]);
    }
  }
  // relu + restage 16x128 bf16 tile into LDS (D-layout -> memory layout)
#pragma unroll
  for (int nt = 0; nt < 8; ++nt) {
    int n = nt * 16 + l16;
    float bias = b1[n];
#pragma unroll
    for (int r = 0; r < 8; ++r) {
      float v = acc[nt][r] + bias;
      v = v > 0.0f ? v : 0.0f;
      T[(r + 8 * half) * C_HID + n] = f2bf(v);
    }
  }
  // ---- GEMM2: t @ W2^T ---- (same-wave LDS RAW; compiler inserts dscnt wait)
  v8f acc2[8] = {};
#pragma unroll
  for (int kk = 0; kk < 4; ++kk) {
    v16bf a;
#pragma unroll
    for (int j = 0; j < 16; ++j) a[j] = T[l16 * C_HID + kk * 32 + kmap(j, half)];
#pragma unroll
    for (int nt = 0; nt < 8; ++nt) {
      const __bf16* brow = sW2 + (nt * 16 + l16) * C_HID;
      v16bf b;
#pragma unroll
      for (int j = 0; j < 16; ++j) b[j] = brow[kk * 32 + kmap(j, half)];
      acc2[nt] = wmma_bf16(a, b, acc2[nt]);
    }
  }
#pragma unroll
  for (int nt = 0; nt < 8; ++nt) {
    int n = nt * 16 + l16;
    float bias = b2[n];
#pragma unroll
    for (int r = 0; r < 8; ++r) {
      float v = acc2[nt][r] + bias;
      if (RELU) v = v > 0.0f ? v : 0.0f;
      out[(size_t)(r0 + r + 8 * half) * C_HID + n] = v;
    }
  }
}

// ---------------------------------------------------------------------------
__global__ void k_pool(const float* __restrict__ h, const int* __restrict__ batch,
                       float* __restrict__ sums, float* __restrict__ cnt) {
  int node = blockIdx.x;
  if (node >= N_NODES) return;
  int g = batch[node];
  int c = threadIdx.x;                 // blockDim.x == 128
  atomicAdd(&sums[(size_t)g * C_HID + c], h[(size_t)node * C_HID + c]);
  if (c == 0) atomicAdd(&cnt[g], 1.0f);
}

__global__ void k_final(const float* __restrict__ sums, const float* __restrict__ cnt,
                        const float* __restrict__ W,    // [16][128]
                        const float* __restrict__ b,    // [16]
                        float* __restrict__ out) {      // [512][16]
  int g = blockIdx.x;
  int o = threadIdx.x;                 // blockDim.x == 16
  float c = cnt[g];
  float inv = 1.0f / (c > 1.0f ? c : 1.0f);
  float acc = b[o];
  for (int k = 0; k < C_HID; ++k)
    acc += sums[(size_t)g * C_HID + k] * inv * W[o * C_HID + k];
  out[(size_t)g * C_OUT + o] = acc;
}

// ---------------------------------------------------------------------------
extern "C" void kernel_launch(void* const* d_in, const int* in_sizes, int n_in,
                              void* d_out, int out_size, void* d_ws, size_t ws_size,
                              hipStream_t stream) {
  (void)in_sizes; (void)n_in; (void)out_size; (void)ws_size;

  // inputs (setup_inputs dict order; params dict flattened in insertion order)
  const float* x       = (const float*)d_in[0];        // [N,64]
  const int*   eidx    = (const int*)d_in[1];          // [2,E]
  const float* eattr   = (const float*)d_in[2];        // [E,64]
  const int*   batch   = (const int*)d_in[3];          // [N]
  const float* We_f    = (const float*)d_in[4];        // [128,64]
  const float* be      = (const float*)d_in[5];
  const float* Wc1_f   = (const float*)d_in[6];        // [64,128]
  const float* bc1     = (const float*)d_in[7];
  const float* W11_f   = (const float*)d_in[8];        // [128,64]
  const float* b11     = (const float*)d_in[9];
  const float* W12_f   = (const float*)d_in[10];       // [128,128]
  const float* b12     = (const float*)d_in[11];
  const float* Wc2_f   = (const float*)d_in[12];       // [128,128]
  const float* bc2     = (const float*)d_in[13];
  const float* W21_f   = (const float*)d_in[14];       // [128,128]
  const float* b21     = (const float*)d_in[15];
  const float* W22_f   = (const float*)d_in[16];       // [128,128]
  const float* b22     = (const float*)d_in[17];
  const float* Wout    = (const float*)d_in[18];       // [16,128]
  const float* bout    = (const float*)d_in[19];
  float* out = (float*)d_out;

  const int* src = eidx;
  const int* dst = eidx + N_EDGES;

  // workspace layout (~282 MB)
  char* ws = (char*)d_ws;
  const size_t EPROJ_B = (size_t)N_EDGES * C_HID * 2;   // 204.8 MB bf16
  const size_t NODE_B  = (size_t)N_NODES * C_HID * 4;   // 25.6 MB fp32
  __bf16* eproj = (__bf16*)ws;
  float*  aggr  = (float*)(ws + EPROJ_B);
  float*  h1    = (float*)(ws + EPROJ_B + NODE_B);
  float*  h2    = (float*)(ws + EPROJ_B + 2 * NODE_B);
  __bf16* wbf   = (__bf16*)(ws + EPROJ_B + 3 * NODE_B); // 90112 elems bf16
  __bf16* We  = wbf + 0;
  __bf16* Wc1 = wbf + 8192;
  __bf16* W11 = wbf + 16384;
  __bf16* W12 = wbf + 24576;
  __bf16* Wc2 = wbf + 40960;
  __bf16* W21 = wbf + 57344;
  __bf16* W22 = wbf + 73728;
  float* sums = (float*)(ws + EPROJ_B + 3 * NODE_B + 90112 * 2);
  float* cnts = sums + (size_t)N_GRAPH * C_HID;

  // 0) weights -> bf16
  k_cvt_bf16<<<(8192 + 255) / 256, 256, 0, stream>>>(We_f, We, 8192);
  k_cvt_bf16<<<(8192 + 255) / 256, 256, 0, stream>>>(Wc1_f, Wc1, 8192);
  k_cvt_bf16<<<(8192 + 255) / 256, 256, 0, stream>>>(W11_f, W11, 8192);
  k_cvt_bf16<<<(16384 + 255) / 256, 256, 0, stream>>>(W12_f, W12, 16384);
  k_cvt_bf16<<<(16384 + 255) / 256, 256, 0, stream>>>(Wc2_f, Wc2, 16384);
  k_cvt_bf16<<<(16384 + 255) / 256, 256, 0, stream>>>(W21_f, W21, 16384);
  k_cvt_bf16<<<(16384 + 255) / 256, 256, 0, stream>>>(W22_f, W22, 16384);

  const int ETILES = N_EDGES / 16;   // 50000
  const int NTILES = N_NODES / 16;   // 3125
  const int WPB = 8;                 // waves per 256-thread block

  // 1) e_proj (shared by both convs) — LDS-staged weights, 1 tile/wave
  k_edge_proj<<<(ETILES + WPB - 1) / WPB, 32 * WPB, 0, stream>>>(
      eattr, We, be, eproj, ETILES);

  // 2) conv1
  k_zero<<<2048, 256, 0, stream>>>(aggr, N_NODES * C_HID);
  k_conv_edge<C_IN><<<(ETILES + WPB - 1) / WPB, 32 * WPB, 0, stream>>>(
      eproj, Wc1, bc1, x, src, dst, aggr, ETILES);
  k_mlp<C_IN, true><<<(NTILES + WPB - 1) / WPB, 32 * WPB, 0, stream>>>(
      x, aggr, W11, b11, W12, b12, h1, NTILES);

  // 3) conv2
  k_zero<<<2048, 256, 0, stream>>>(aggr, N_NODES * C_HID);
  k_conv_edge<C_HID><<<(ETILES + WPB - 1) / WPB, 32 * WPB, 0, stream>>>(
      eproj, Wc2, bc2, h1, src, dst, aggr, ETILES);
  k_mlp<C_HID, false><<<(NTILES + WPB - 1) / WPB, 32 * WPB, 0, stream>>>(
      h1, aggr, W21, b21, W22, b22, h2, NTILES);

  // 4) mean pool + head
  k_zero<<<256, 256, 0, stream>>>(sums, N_GRAPH * C_HID + N_GRAPH);
  k_pool<<<N_NODES, C_HID, 0, stream>>>(h2, batch, sums, cnts);
  k_final<<<N_GRAPH, C_OUT, 0, stream>>>(sums, cnts, Wout, bout, out);
}
